// Spectral_MP_GNN_68564857914141
// MI455X (gfx1250) — matrix operand order repaired and checked by reading, not compiled, not run on hardware
//
#include <hip/hip_runtime.h>
#include <hip/hip_bf16.h>

typedef __attribute__((ext_vector_type(16))) _Float16 v16h;
typedef __attribute__((ext_vector_type(8)))  float    v8f;

#define HDIM 128
#define NN   65536
#define EE   262144
#define ESPCN 16384
#define GG   4
#define KKE  64
#define NG   (NN/GG)
#define GKN  (GG*KKE)
#define LL   2
#define EPSV 1e-5f

// ---------------- WMMA fragment helpers (wave32, 16x16x32 f16) ----------------
// A (16x32 f16) from LDS row-major tile: lane m=l&15, kbase=(l>>4)*8;
// elements 0..7 -> K=kbase+e, elements 8..15 -> K=kbase+16+(e-8).
__device__ inline v16h load_A_frag_lds(const _Float16* base, int stride, int lane) {
  int m  = lane & 15;
  int kb = (lane >> 4) << 3;
  const _Float16* p = base + m * stride + kb;
  v16h a;
#pragma unroll
  for (int e = 0; e < 8; ++e) a[e] = p[e];
#pragma unroll
  for (int e = 0; e < 8; ++e) a[8 + e] = p[16 + e];
  return a;
}

// B (32x16 f16) from global pre-transposed Bt[n][k] (row length ldbt):
// lane n=l&15, kbase=(l>>4)*8; elements 0..7 -> K=kbase+e, elements 8..15 -> K=kbase+16+(e-8).
__device__ inline v16h load_B_frag_g(const _Float16* Bt, int ldbt, int n0, int k0, int lane) {
  int n  = n0 + (lane & 15);
  int kb = k0 + ((lane >> 4) << 3);
  const _Float16* p = Bt + (size_t)n * ldbt + kb;
  v16h b;
#pragma unroll
  for (int e = 0; e < 8; ++e) b[e] = p[e];
#pragma unroll
  for (int e = 0; e < 8; ++e) b[8 + e] = p[16 + e];
  return b;
}

__device__ inline v8f zero8() {
  v8f z = {0.f, 0.f, 0.f, 0.f, 0.f, 0.f, 0.f, 0.f};
  return z;
}

// ---------------- f32 -> f16 transpose (zero-padded rows) ----------------
// out[c*Rpad + r] = in[r*ldin + c] for r < R else 0.
__global__ void tr_f32_f16(const float* __restrict__ in, int R, int C, int ldin,
                           _Float16* __restrict__ out, int Rpad) {
  int c = blockIdx.y;
  for (int r = blockIdx.x * blockDim.x + threadIdx.x; r < Rpad; r += gridDim.x * blockDim.x) {
    float v = (r < R) ? in[(size_t)r * ldin + c] : 0.f;
    out[(size_t)c * Rpad + r] = (_Float16)v;
  }
}

__global__ void zero_f32(float* p, size_t n) {
  size_t i = (size_t)blockIdx.x * blockDim.x + threadIdx.x;
  if (i < n) p[i] = 0.f;
}

// ---------------- Fused encoder: LN(relu(X@W1+b1)@W2+b2)*g+b (+resid, +scatter) -------
// X rows gathered on the fly from up to 3 segments. 16 rows x 128 cols per block.
__global__ void __launch_bounds__(256)
enc_kernel(const float* __restrict__ seg0, int w0, int ld0, const int* __restrict__ idx0,
           const float* __restrict__ seg1, int w1c, int ld1, const int* __restrict__ idx1,
           const float* __restrict__ seg2, int w2c, int ld2,
           int M, int din, int dinpad,
           const _Float16* __restrict__ W1t, const float* __restrict__ b1v,
           const _Float16* __restrict__ W2t, const float* __restrict__ b2v,
           const float* __restrict__ gam, const float* __restrict__ bet,
           const float* __restrict__ resid, int ldres,
           float* __restrict__ out, int ldout,
           float* __restrict__ agg, const int* __restrict__ aggidx) {
  __shared__ _Float16 Xt[16 * 640];
  __shared__ _Float16 H1s[16 * HDIM];
  __shared__ float    H2s[16 * HDIM];
  __shared__ float    mu_s[16], rs_s[16];

  int tid  = threadIdx.x;
  int row0 = blockIdx.x * 16;

  // Stage input tile (gather + f16 convert + zero pad)
  for (int t = tid; t < 16 * dinpad; t += 256) {
    int r = t / dinpad, c = t - r * dinpad;
    int row = row0 + r;
    float v = 0.f;
    if (row < M && c < din) {
      if (c < w0) {
        int rr = idx0 ? idx0[row] : row;
        v = seg0[(size_t)rr * ld0 + c];
      } else if (c < w0 + w1c) {
        int rr = idx1 ? idx1[row] : row;
        v = seg1[(size_t)rr * ld1 + (c - w0)];
      } else {
        v = seg2[(size_t)row * ld2 + (c - w0 - w1c)];
      }
    }
    Xt[r * dinpad + c] = (_Float16)v;
  }
  __syncthreads();

  int wave = tid >> 5, lane = tid & 31;
  int n0 = wave * 16;

  // GEMM1: H1 = relu(X @ W1 + b1)
  v8f acc = zero8();
  for (int k0 = 0; k0 < dinpad; k0 += 32) {
    v16h a = load_A_frag_lds(Xt + k0, dinpad, lane);
    v16h b = load_B_frag_g(W1t, dinpad, n0, k0, lane);
    acc = __builtin_amdgcn_wmma_f32_16x16x32_f16(false, a, false, b, (short)0, acc, false, false);
  }
  {
    int n = n0 + (lane & 15);
    float bb = b1v[n];
    int mb = (lane >> 4) << 3;
#pragma unroll
    for (int v = 0; v < 8; ++v) {
      float h = acc[v] + bb;
      H1s[(mb + v) * HDIM + n] = (_Float16)(h > 0.f ? h : 0.f);
    }
  }
  __syncthreads();

  // GEMM2: H2 = H1 @ W2 + b2
  v8f acc2 = zero8();
  for (int k0 = 0; k0 < HDIM; k0 += 32) {
    v16h a = load_A_frag_lds(H1s + k0, HDIM, lane);
    v16h b = load_B_frag_g(W2t, HDIM, n0, k0, lane);
    acc2 = __builtin_amdgcn_wmma_f32_16x16x32_f16(false, a, false, b, (short)0, acc2, false, false);
  }
  {
    int n = n0 + (lane & 15);
    float bb = b2v[n];
    int mb = (lane >> 4) << 3;
#pragma unroll
    for (int v = 0; v < 8; ++v) H2s[(mb + v) * HDIM + n] = acc2[v] + bb;
  }
  __syncthreads();

  // LayerNorm statistics per row
  if (tid < 16) {
    float s = 0.f, sq = 0.f;
    for (int c = 0; c < HDIM; ++c) { float h = H2s[tid * HDIM + c]; s += h; sq += h * h; }
    float m  = s / HDIM;
    float var = sq / HDIM - m * m;
    mu_s[tid] = m;
    rs_s[tid] = rsqrtf(var + EPSV);
  }
  __syncthreads();

  // Apply LN, residual, write out, optional atomic scatter-add (segment_sum)
  for (int t = tid; t < 16 * HDIM; t += 256) {
    int r = t >> 7, c = t & 127;
    int row = row0 + r;
    if (row < M) {
      float v = (H2s[t] - mu_s[r]) * rs_s[r] * gam[c] + bet[c];
      if (resid) v += resid[(size_t)row * ldres + c];
      out[(size_t)row * ldout + c] = v;
      if (agg) atomicAdd(agg + (size_t)aggidx[row] * HDIM + c, v);
    }
  }
}

// ---------------- Generic WMMA GEMM: C = op(A) @ Bt^T (+bias, relu) --------------
// A f32 (optionally transposed access), Bt is N x K f16 row-major; K % 32 == 0.
// 16 rows x 128 cols per block, 8 waves.
__global__ void __launch_bounds__(256)
gemm_kernel(const float* __restrict__ A, int lda, int transA, int M, int K,
            const _Float16* __restrict__ Bt, int ldbt,
            const float* __restrict__ bias, int dorelu,
            float* __restrict__ C, int ldc) {
  __shared__ _Float16 At[16 * 32];
  int tid = threadIdx.x;
  int row0 = blockIdx.x * 16;
  int wave = tid >> 5, lane = tid & 31;
  int n0 = wave * 16;

  v8f acc = zero8();
  for (int k0 = 0; k0 < K; k0 += 32) {
    __syncthreads();
    for (int t = tid; t < 512; t += 256) {
      int r = t >> 5, c = t & 31;
      int m = row0 + r, k = k0 + c;
      float v = 0.f;
      if (m < M) v = transA ? A[(size_t)k * lda + m] : A[(size_t)m * lda + k];
      At[r * 32 + c] = (_Float16)v;
    }
    __syncthreads();
    v16h a = load_A_frag_lds(At, 32, lane);
    v16h b = load_B_frag_g(Bt, ldbt, n0, k0, lane);
    acc = __builtin_amdgcn_wmma_f32_16x16x32_f16(false, a, false, b, (short)0, acc, false, false);
  }

  int n = n0 + (lane & 15);
  float bb = bias ? bias[n] : 0.f;
  int mb = (lane >> 4) << 3;
#pragma unroll
  for (int v = 0; v < 8; ++v) {
    int row = row0 + mb + v;
    if (row < M) {
      float x = acc[v] + bb;
      if (dorelu && x < 0.f) x = 0.f;
      C[(size_t)row * ldc + n] = x;
    }
  }
}

// ---------------- Decoder tail: out = relu_h @ W2(128x3) + b2 ----------------
__global__ void dec2_kernel(const float* __restrict__ h, const float* __restrict__ w2,
                            const float* __restrict__ b2, float* __restrict__ out, int M) {
  int r = blockIdx.x * blockDim.x + threadIdx.x;
  if (r >= M) return;
  float a0 = b2[0], a1 = b2[1], a2 = b2[2];
  const float* hr = h + (size_t)r * HDIM;
  for (int c = 0; c < HDIM; ++c) {
    float hv = hr[c];
    a0 += hv * w2[c * 3 + 0];
    a1 += hv * w2[c * 3 + 1];
    a2 += hv * w2[c * 3 + 2];
  }
  out[r * 3 + 0] = a0; out[r * 3 + 1] = a1; out[r * 3 + 2] = a2;
}

// =============================== host orchestration ===============================
extern "C" void kernel_launch(void* const* d_in, const int* in_sizes, int n_in,
                              void* d_out, int out_size, void* d_ws, size_t ws_size,
                              hipStream_t stream) {
  (void)in_sizes; (void)n_in; (void)out_size; (void)ws_size;

  // Input order: setup_inputs() dict insertion order, params flattened recursively
  // (each encoder dict: w1,b1,w2,b2,g,b).
  const float* in_x   = (const float*)d_in[0];
  const float* in_ea  = (const float*)d_in[1];
  const float* in_ext = (const float*)d_in[2];
  const float* in_spc = (const float*)d_in[3];
  const float* in_sea = (const float*)d_in[4];
  const float* in_V   = (const float*)d_in[5];
  const int*   ei     = (const int*)d_in[82];
  const int*   sei    = (const int*)d_in[83];
  const int *ei0 = ei, *ei1 = ei + EE;
  const int *sei0 = sei, *sei1 = sei + ESPCN;

  auto P = [&](int i) -> const float* { return (const float*)d_in[i]; };
  // encoder blocks: 0 node_enc,1 edge_enc,2 spc_node_enc,3 spc_edge_enc,
  // 4/5 proc0 edge/node, 6/7 proc1 edge/node, 8/9 sproc0, 10/11 sproc1
  int encBase[12] = {6, 12, 18, 24, 30, 36, 42, 48, 54, 60, 66, 72};
  int encDin[12]  = {6, 4, 3, 3, 640, 387, 640, 387, 640, 384, 640, 384};
  int encPad[12]  = {32, 32, 32, 32, 640, 416, 640, 416, 640, 384, 640, 384};
  const int DEC_W1 = 78, DEC_B1 = 79, DEC_W2 = 80, DEC_B2 = 81;

  // ---- workspace bump allocator ----
  char* ws = (char*)d_ws;
  size_t off = 0;
  auto alloc = [&](size_t bytes) -> void* {
    void* p = ws + off;
    off = (off + bytes + 255) & ~(size_t)255;
    return p;
  };
  float* xA   = (float*)alloc((size_t)NN * 256 * 4);
  float* xB   = (float*)alloc((size_t)NN * 256 * 4);
  float* eaA  = (float*)alloc((size_t)EE * 128 * 4);
  float* eaB  = (float*)alloc((size_t)EE * 128 * 4);
  float* seaA = (float*)alloc((size_t)ESPCN * 128 * 4);
  float* seaB = (float*)alloc((size_t)ESPCN * 128 * 4);
  float* spcA = (float*)alloc((size_t)GKN * 256 * 4);
  float* spcB = (float*)alloc((size_t)GKN * 256 * 4);
  float* agg  = (float*)alloc((size_t)NN * 128 * 4);
  float* sagg = (float*)alloc((size_t)GKN * 128 * 4);
  float* dech = (float*)alloc((size_t)NN * 128 * 4);
  _Float16* w1t[12];
  _Float16* w2t[12];
  for (int i = 0; i < 12; ++i) {
    w1t[i] = (_Float16*)alloc((size_t)128 * encPad[i] * 2);
    w2t[i] = (_Float16*)alloc((size_t)128 * 128 * 2);
  }
  _Float16* decw1t = (_Float16*)alloc((size_t)128 * 256 * 2);
  _Float16* spcBt  = (_Float16*)alloc((size_t)GG * 128 * KKE * 2);
  _Float16* xBt    = (_Float16*)alloc((size_t)GG * 128 * NG * 2);

  // ---- weight conversion (f32 -> transposed, zero-padded f16 panels) ----
  for (int i = 0; i < 12; ++i) {
    dim3 g1((encPad[i] + 255) / 256, 128);
    tr_f32_f16<<<g1, 256, 0, stream>>>(P(encBase[i] + 0), encDin[i], 128, 128, w1t[i], encPad[i]);
    tr_f32_f16<<<dim3(1, 128), 256, 0, stream>>>(P(encBase[i] + 2), 128, 128, 128, w2t[i], 128);
  }
  tr_f32_f16<<<dim3(1, 128), 256, 0, stream>>>(P(DEC_W1), 256, 128, 128, decw1t, 256);

  auto runEnc = [&](int e, int M,
                    const float* s0, int w0, int ld0, const int* i0,
                    const float* s1, int w1c, int ld1, const int* i1,
                    const float* s2, int w2c, int ld2,
                    const float* resid, int ldres,
                    float* out, int ldout, float* ag, const int* agidx) {
    int b = encBase[e];
    enc_kernel<<<(M + 15) / 16, 256, 0, stream>>>(
        s0, w0, ld0, i0, s1, w1c, ld1, i1, s2, w2c, ld2,
        M, encDin[e], encPad[e],
        w1t[e], P(b + 1), w2t[e], P(b + 3), P(b + 4), P(b + 5),
        resid, ldres, out, ldout, ag, agidx);
  };

  auto eigBroadcast = [&](const float* spcCur, float* xDst) {
    for (int g = 0; g < GG; ++g) {
      _Float16* bt = spcBt + (size_t)g * 128 * KKE;
      tr_f32_f16<<<dim3(1, 128), 256, 0, stream>>>(spcCur + (size_t)g * KKE * 256, KKE, 128, 256, bt, KKE);
      gemm_kernel<<<NG / 16, 256, 0, stream>>>(in_V + (size_t)g * NG * KKE, KKE, 0, NG, KKE,
                                               bt, KKE, nullptr, 0,
                                               xDst + (size_t)g * NG * 256 + 128, 256);
    }
  };
  auto eigPool = [&](const float* xCur, float* spcDst) {
    for (int g = 0; g < GG; ++g) {
      _Float16* bt = xBt + (size_t)g * 128 * NG;
      tr_f32_f16<<<dim3((NG + 255) / 256, 128), 256, 0, stream>>>(xCur + (size_t)g * NG * 256, NG, 128, 256, bt, NG);
      gemm_kernel<<<(KKE + 15) / 16, 256, 0, stream>>>(in_V + (size_t)g * NG * KKE, KKE, 1, KKE, NG,
                                                       bt, NG, nullptr, 0,
                                                       spcDst + (size_t)g * KKE * 256 + 128, 256);
    }
  };

  // ---- initial encoders ----
  runEnc(0, NN,    in_x, 6, 6, nullptr,   nullptr, 0, 0, nullptr,  nullptr, 0, 0,
         nullptr, 0, xA, 256, nullptr, nullptr);
  runEnc(1, EE,    in_ea, 4, 4, nullptr,  nullptr, 0, 0, nullptr,  nullptr, 0, 0,
         nullptr, 0, eaA, 128, nullptr, nullptr);
  runEnc(2, GKN,   in_spc, 3, 3, nullptr, nullptr, 0, 0, nullptr,  nullptr, 0, 0,
         nullptr, 0, spcA, 256, nullptr, nullptr);
  runEnc(3, ESPCN, in_sea, 3, 3, nullptr, nullptr, 0, 0, nullptr,  nullptr, 0, 0,
         nullptr, 0, seaA, 128, nullptr, nullptr);
  // x = [x_enc | eig_broadcast(V, spc_enc)],  spc = [spc_enc | eig_pool(V, x_enc)]
  eigBroadcast(spcA, xA);
  eigPool(xA, spcA);

  float *xCur = xA, *xNxt = xB, *eaCur = eaA, *eaNxt = eaB;
  float *seaCur = seaA, *seaNxt = seaB, *spcCur = spcA, *spcNxt = spcB;

  for (int l = 0; l < LL; ++l) {
    int pe = 4 + 2 * l, pn = 5 + 2 * l, se = 8 + 2 * l, sn = 9 + 2 * l;
    zero_f32<<<(int)(((size_t)NN * 128 + 255) / 256), 256, 0, stream>>>(agg, (size_t)NN * 128);
    zero_f32<<<(int)(((size_t)GKN * 128 + 255) / 256), 256, 0, stream>>>(sagg, (size_t)GKN * 128);

    // edge MLP: ue = enc([x[dst], x[src], ea]) + ea ; agg = segment_sum(ue, src)
    runEnc(pe, EE, xCur, 256, 256, ei1,  xCur, 256, 256, ei0,  eaCur, 128, 128,
           eaCur, 128, eaNxt, 128, agg, ei0);
    runEnc(se, ESPCN, spcCur, 256, 256, sei1,  spcCur, 256, 256, sei0,  seaCur, 128, 128,
           seaCur, 128, seaNxt, 128, sagg, sei0);
    // node MLP: x_i = x[:, :128] + enc([x, agg, ext])
    runEnc(pn, NN, xCur, 256, 256, nullptr,  agg, 128, 128, nullptr,  in_ext, 3, 3,
           xCur, 256, xNxt, 256, nullptr, nullptr);
    runEnc(sn, GKN, spcCur, 256, 256, nullptr,  sagg, 128, 128, nullptr,  nullptr, 0, 0,
           spcCur, 256, spcNxt, 256, nullptr, nullptr);
    // spectral exchange uses truncated OLD states
    eigBroadcast(spcCur, xNxt);
    eigPool(xCur, spcNxt);

    float* t;
    t = xCur; xCur = xNxt; xNxt = t;
    t = eaCur; eaCur = eaNxt; eaNxt = t;
    t = seaCur; seaCur = seaNxt; seaNxt = t;
    t = spcCur; spcCur = spcNxt; spcNxt = t;
  }

  // ---- decoder ----
  gemm_kernel<<<NN / 16, 256, 0, stream>>>(xCur, 256, 0, NN, 256, decw1t, 256,
                                           P(DEC_B1), 1, dech, 128);
  dec2_kernel<<<(NN + 255) / 256, 256, 0, stream>>>(dech, P(DEC_W2), P(DEC_B2),
                                                    (float*)d_out, NN);
}
